// GPT_4930622456357
// MI455X (gfx1250) — compile-verified, matrix operand order checked by reading, not run on hardware
//
#include <hip/hip_runtime.h>
#include <hip/hip_bf16.h>
#include <math.h>

// ---------------------------------------------------------------------------
// Decision Transformer forward for MI455X (gfx1250), wave32 + WMMA.
// - All GEMM-shaped work through v_wmma_f32_16x16x32_f16 (f32 accumulate).
// - Weights pre-packed once into f16 LDS-image panels; per-block staging uses
//   GLOBAL_LOAD_ASYNC_TO_LDS_B128 + s_wait_asynccnt (CDNA5 async path).
// - Activations feeding WMMA kept in f16; A fragments = 2 aligned b128 loads.
// - B fragments read from LDS with ds_load_b128 (padded rows, no conflicts).
// ---------------------------------------------------------------------------

typedef _Float16 halft;
typedef __attribute__((ext_vector_type(16))) _Float16 v16h;
typedef __attribute__((ext_vector_type(8)))  _Float16 v8h;
typedef __attribute__((ext_vector_type(4)))  _Float16 v4h;
typedef __attribute__((ext_vector_type(8)))  float    v8f;

#define DEV __device__ __forceinline__

constexpr int B_ = 32, K_ = 64, D_ = 128, V_ = 18, L_ = 6, H_ = 8;
constexpr int T_ = 3 * K_ - 1;        // 191
constexpr int DH_ = D_ / H_;          // 16
constexpr float EPS_ = 1e-5f;

DEV int lane_id() { return threadIdx.x & 31; }
DEV int wave_id() { return (int)(threadIdx.x >> 5); }

DEV v8f wmma_f16(v16h a, v16h b, v8f c) {
  return __builtin_amdgcn_wmma_f32_16x16x32_f16(
      false, a, false, b, (short)0, c, false, false);
}

// Fragment k map (CDNA5 ISA 7.12.2): lanes 0-15 hold K {0..7,16..23};
// lanes 16-31 hold K {8..15,24..31} => two contiguous 8-runs per lane.
DEV v16h frag_from_h16(const halft* p, int k0, int khalf) {
  v8h r0 = *(const v8h*)(p + k0 + khalf);        // 16B load
  v8h r1 = *(const v8h*)(p + k0 + 16 + khalf);   // 16B load
  return __builtin_shufflevector(r0, r1, 0,1,2,3,4,5,6,7,8,9,10,11,12,13,14,15);
}

// Async 16B global->LDS copy (per active lane), ASYNCcnt-tracked.
DEV void async_copy_b128(unsigned lds_byte_addr, const void* gsrc) {
  asm volatile("global_load_async_to_lds_b128 %0, %1, off"
               :: "v"(lds_byte_addr), "v"(gsrc) : "memory");
}
DEV void async_wait0() { asm volatile("s_wait_asynccnt 0x0" ::: "memory"); }

// ---------------------------------------------------------------------------
// Weight packing into LDS panel images: [ntile][chunk][n(16)][k(BSTR)], f16,
// +8-half pad per row (bank rotation), zero-filled pad & out-of-Ncols cols.
// ---------------------------------------------------------------------------
// transpose-pack: W is [KDIM x ldb] row-major (GEMM weights, k-major)
__global__ void pack_weights_t(const float* __restrict__ W,
                               halft* __restrict__ P,
                               int KC, int BSTR, int chunks, int ldb,
                               int Ncols, int Ntiles) {
  int idx = blockIdx.x * blockDim.x + threadIdx.x;
  int total = Ntiles * chunks * 16 * BSTR;
  if (idx >= total) return;
  int k  = idx % BSTR;
  int n  = (idx / BSTR) & 15;
  int ch = (idx / (BSTR * 16)) % chunks;
  int nt = idx / (BSTR * 16 * chunks);
  int col = nt * 16 + n;
  float v = (k < KC && col < Ncols) ? W[(ch * KC + k) * ldb + col] : 0.f;
  P[idx] = (halft)v;
}

// direct-pack: W is [OC x KDIM] row-major (conv OIHW weights)
__global__ void pack_weights_d(const float* __restrict__ W,
                               halft* __restrict__ P,
                               int KDIM, int BSTR, int OC) {
  int idx = blockIdx.x * blockDim.x + threadIdx.x;
  if (idx >= (OC / 16) * 16 * BSTR) return;
  int k = idx % BSTR;
  int oc = idx / BSTR;
  P[idx] = (halft)((k < KDIM) ? W[oc * KDIM + k] : 0.f);
}

// ---------------------------------------------------------------------------
// Implicit-GEMM convolution, compile-time geometry, ReLU, f16 output.
// Block: 4 waves, one 16-channel N-tile, 16 output-pixel M-tiles.
// Weight panel staged via async global->LDS copy.
// ---------------------------------------------------------------------------
template <typename TIN, int IC, int IH, int KH, int STRIDE, int OC, int OH>
__global__ void conv_implicit_gemm(const TIN* __restrict__ in,
                                   const halft* __restrict__ wp,
                                   const float* __restrict__ bias,
                                   halft* __restrict__ out, int NIMG) {
  constexpr int IW = IH, OW = OH, KW = KH;
  constexpr int PIX = OH * OW;
  constexpr int KDIM = IC * KH * KW;
  constexpr int NT = OC / 16;
  constexpr int BSTR = KDIM + 8;           // padded LDS row (bank rotate)
  __shared__ halft wpanel[16 * BSTR];      // [n][k]

  const int Mtiles = (NIMG * PIX) >> 4;
  int nt = blockIdx.x % NT;
  int bm = blockIdx.x / NT;
  int n0 = nt * 16;

  {  // async-stage the pre-packed panel (16*BSTR halfs = 2*BSTR 16B chunks)
    const halft* src = wp + nt * 16 * BSTR;
    unsigned lds0 = (unsigned)(uintptr_t)(&wpanel[0]);
    for (int c = threadIdx.x; c < 2 * BSTR; c += blockDim.x)
      async_copy_b128(lds0 + c * 16, src + c * 8);
    async_wait0();
  }
  __syncthreads();

  int lane = lane_id(), wv = wave_id();
  int khalf = (lane >= 16) ? 8 : 0;
  const halft* bp = wpanel + (lane & 15) * BSTR;
  int oc = n0 + (lane & 15);
  float bval = bias[oc];

  const TIN* inA[4];
  bool mok[4];
  v8f acc[4];
#pragma unroll
  for (int i = 0; i < 4; ++i) {
    int mt = bm * 16 + wv + 4 * i;
    mok[i] = (mt < Mtiles);
    int mA = (mok[i] ? mt : 0) * 16 + (lane & 15);
    int img = mA / PIX, p = mA % PIX;
    int oh = p / OW, ow = p % OW;
    inA[i] = in + img * IC * IH * IW + (oh * STRIDE) * IW + ow * STRIDE;
    acc[i] = (v8f){};
  }

  for (int k0 = 0; k0 < KDIM; k0 += 32) {
    v16h bfrag = frag_from_h16(bp, k0, khalf);
#pragma unroll
    for (int i = 0; i < 4; ++i) {
      v16h a;
      if constexpr (KW == 8) {
        // each 8-run == one filter row -> aligned float4 pairs (conv1, f32 in)
        float seg[16];
#pragma unroll
        for (int r = 0; r < 2; ++r) {
          int kb = k0 + khalf + r * 16;
          int ic = kb / (KH * KW);
          int kh = (kb % (KH * KW)) / KW;  // kw == 0 by construction
          const float4* p = (const float4*)(inA[i] + (ic * IH + kh) * IW);
          float4 f0 = p[0], f1 = p[1];
          seg[r*8+0]=f0.x; seg[r*8+1]=f0.y; seg[r*8+2]=f0.z; seg[r*8+3]=f0.w;
          seg[r*8+4]=f1.x; seg[r*8+5]=f1.y; seg[r*8+6]=f1.z; seg[r*8+7]=f1.w;
        }
#pragma unroll
        for (int j = 0; j < 16; ++j) a[j] = (halft)seg[j];
      } else {
#pragma unroll
        for (int j = 0; j < 16; ++j) {
          int k = k0 + ((j < 8) ? (j + khalf) : (j + 8 + khalf));
          int ic = k / (KH * KW);
          int r  = k % (KH * KW);
          a[j] = (halft)inA[i][(ic * IH + r / KW) * IW + (r % KW)];
        }
      }
      acc[i] = wmma_f16(a, bfrag, acc[i]);
    }
  }

#pragma unroll
  for (int i = 0; i < 4; ++i) {
    if (!mok[i]) continue;
    int m0 = (bm * 16 + wv + 4 * i) * 16;
#pragma unroll
    for (int r = 0; r < 8; ++r) {
      int m = m0 + r + ((lane >= 16) ? 8 : 0);
      int img = m / PIX, p = m % PIX;
      float v = acc[i][r] + bval;
      v = v > 0.f ? v : 0.f;
      out[(img * OC + oc) * PIX + p] = (halft)v;
    }
  }
}

// ---------------------------------------------------------------------------
// Generic WMMA GEMM, f16 A, async-staged pre-packed f16 B panels.
// MODE: 0 bias->f16 | 2 bias+tanh->f32 | 3 bias+GELU->f16
//       4 bias+residual->f32 | 5 head (state-token remap, Ncols guard)->f32
// ---------------------------------------------------------------------------
template <int MODE, int LDA, int KDIM>
__global__ void gemm_wmma(const halft* __restrict__ A,
                          const halft* __restrict__ Bp,
                          const float* __restrict__ bias,
                          const float* __restrict__ resid,
                          void* __restrict__ Cv, int ldc,
                          int M, int N, int Ncols) {
  constexpr bool HOUT = (MODE == 0 || MODE == 3);
  constexpr int KC = (KDIM > 1568) ? 1568 : KDIM;  // KDIM % KC == 0
  constexpr int CHUNKS = KDIM / KC;
  constexpr int BSTR = KC + 8;
  __shared__ halft bpanel[16 * BSTR];

  const int Ntiles = N >> 4;
  const int Mtiles = M >> 4;
  int nt = blockIdx.x % Ntiles;
  int bm = blockIdx.x / Ntiles;
  int n0 = nt * 16;

  int lane = lane_id(), wv = wave_id();
  int khalf = (lane >= 16) ? 8 : 0;
  const halft* bp = bpanel + (lane & 15) * BSTR;
  int col = n0 + (lane & 15);
  bool colok = col < Ncols;
  unsigned lds0 = (unsigned)(uintptr_t)(&bpanel[0]);

  const halft* Ar[4];
  bool mok[4];
  v8f acc[4];
#pragma unroll
  for (int i = 0; i < 4; ++i) {
    int mt = bm * 16 + wv + 4 * i;
    mok[i] = (mt < Mtiles);
    int mA = (mok[i] ? mt : 0) * 16 + (lane & 15);
    int arow = (MODE == 5) ? (mA / K_) * T_ + 3 * (mA % K_) + 1 : mA;
    Ar[i] = A + arow * LDA;
    acc[i] = (v8f){};
  }

  for (int ch = 0; ch < CHUNKS; ++ch) {
    __syncthreads();
    const halft* src = Bp + (nt * CHUNKS + ch) * 16 * BSTR;
    for (int c = threadIdx.x; c < 2 * BSTR; c += blockDim.x)
      async_copy_b128(lds0 + c * 16, src + c * 8);
    async_wait0();
    __syncthreads();

    const int kc = ch * KC;
    for (int k0 = 0; k0 < KC; k0 += 32) {
      v16h b = frag_from_h16(bp, k0, khalf);
#pragma unroll
      for (int i = 0; i < 4; ++i) {
        v16h a = frag_from_h16(Ar[i] + kc, k0, khalf);
        acc[i] = wmma_f16(a, b, acc[i]);
      }
    }
  }

  if (!colok) return;
  float bval = (MODE == 5) ? 0.f : bias[col];
#pragma unroll
  for (int i = 0; i < 4; ++i) {
    if (!mok[i]) continue;
    int m0 = (bm * 16 + wv + 4 * i) * 16 + ((lane >= 16) ? 8 : 0);
    int idx = m0 * ldc + col;
#pragma unroll
    for (int r = 0; r < 8; ++r) {
      float v = acc[i][r] + bval;
      if (MODE == 2) v = tanhf(v);
      if (MODE == 3) v = 0.5f * v * (1.f + erff(v * 0.70710678118654752f));
      if (MODE == 4) v += resid[idx];
      if (HOUT) ((halft*)Cv)[idx] = (halft)v;
      else      ((float*)Cv)[idx] = v;
      idx += ldc;
    }
  }
}

// ---------------------------------------------------------------------------
// LayerNorm over D=128, one wave32 per row; f32 in, f16 out.
// ---------------------------------------------------------------------------
__global__ void layernorm_kernel(const float* __restrict__ x,
                                 const float* __restrict__ g,
                                 const float* __restrict__ b,
                                 halft* __restrict__ y, int rows) {
  int row = blockIdx.x * (blockDim.x >> 5) + wave_id();
  if (row >= rows) return;
  int lane = lane_id();
  const float4 v = ((const float4*)(x + row * D_))[lane];
  float s = v.x + v.y + v.z + v.w;
#pragma unroll
  for (int o = 16; o > 0; o >>= 1) s += __shfl_xor(s, o, 32);
  float mean = s * (1.f / 128.f);
  float4 d = {v.x - mean, v.y - mean, v.z - mean, v.w - mean};
  float q = d.x * d.x + d.y * d.y + d.z * d.z + d.w * d.w;
#pragma unroll
  for (int o = 16; o > 0; o >>= 1) q += __shfl_xor(q, o, 32);
  float rstd = rsqrtf(q * (1.f / 128.f) + EPS_);
  float4 gg = ((const float4*)g)[lane];
  float4 bb = ((const float4*)b)[lane];
  v4h o4;
  o4[0] = (halft)(d.x * rstd * gg.x + bb.x);
  o4[1] = (halft)(d.y * rstd * gg.y + bb.y);
  o4[2] = (halft)(d.z * rstd * gg.z + bb.z);
  o4[3] = (halft)(d.w * rstd * gg.w + bb.w);
  ((v4h*)(y + row * D_))[lane] = o4;
}

// ---------------------------------------------------------------------------
// Token assembly (f32 x).
// ---------------------------------------------------------------------------
__global__ void build_tokens(const float* __restrict__ state_emb,
                             const int* __restrict__ actions,
                             const float* __restrict__ rtgs,
                             const int* __restrict__ timesteps,
                             const float* __restrict__ ret_w,
                             const float* __restrict__ ret_b,
                             const float* __restrict__ act_emb,
                             const float* __restrict__ pos_emb,
                             const float* __restrict__ gpos,
                             float* __restrict__ x) {
  int idx = blockIdx.x * blockDim.x + threadIdx.x;
  if (idx >= B_ * T_ * D_) return;
  int d = idx % D_;
  int t = (idx / D_) % T_;
  int b = idx / (D_ * T_);
  int k = t / 3, ph = t % 3;
  float tok;
  if (ph == 0)      tok = tanhf(rtgs[b * K_ + k] * ret_w[d] + ret_b[d]);
  else if (ph == 1) tok = state_emb[(b * K_ + k) * D_ + d];
  else              tok = tanhf(act_emb[actions[b * K_ + k + 1] * D_ + d]);
  x[idx] = tok + gpos[timesteps[b] * D_ + d] + pos_emb[t * D_ + d];
}

// ---------------------------------------------------------------------------
// Attention scores: S = (Q K^T)/4; f16 Q/K, dh=16 zero-padded to K=32.
// ---------------------------------------------------------------------------
__global__ void attn_scores(const halft* __restrict__ q,
                            const halft* __restrict__ km,
                            float* __restrict__ att) {
  constexpr int TT = (T_ + 15) / 16;  // 12
  int gid = blockIdx.x * (blockDim.x >> 5) + wave_id();
  if (gid >= B_ * H_ * TT * TT) return;
  int nt = gid % TT, mt = (gid / TT) % TT, bh = gid / (TT * TT);
  int b = bh / H_, h = bh % H_;
  int lane = lane_id();
  int khalf = (lane >= 16) ? 8 : 0;
  int t = mt * 16 + (lane & 15);
  int s = nt * 16 + (lane & 15);

  v16h a = {}, bm = {};
  if (t < T_) {
    v8h r = *(const v8h*)(q + (b * T_ + t) * D_ + h * DH_ + khalf);
#pragma unroll
    for (int j = 0; j < 8; ++j) a[j] = r[j];
  }
  if (s < T_) {
    v8h r = *(const v8h*)(km + (b * T_ + s) * D_ + h * DH_ + khalf);
#pragma unroll
    for (int j = 0; j < 8; ++j) bm[j] = r[j];
  }
  v8f acc = {};
  acc = wmma_f16(a, bm, acc);

  int tt0 = mt * 16 + ((lane >= 16) ? 8 : 0);
  int idx = (bh * T_ + tt0) * T_ + s;
  if (mt < TT - 1 && nt < TT - 1) {  // full tile: unguarded streaming stores
#pragma unroll
    for (int r = 0; r < 8; ++r) { att[idx] = acc[r] * 0.25f; idx += T_; }
  } else if (s < T_) {
#pragma unroll
    for (int r = 0; r < 8; ++r) {
      if (tt0 + r < T_) att[idx] = acc[r] * 0.25f;
      idx += T_;
    }
  }
}

// ---------------------------------------------------------------------------
// Causal softmax, one wave32 per (b,h,t) row. Masked entries -> exact 0.
// ---------------------------------------------------------------------------
__global__ void attn_softmax(float* __restrict__ att) {
  int rid = blockIdx.x * (blockDim.x >> 5) + wave_id();
  if (rid >= B_ * H_ * T_) return;
  int t = rid % T_;
  float* row = att + (rid / T_) * (T_ * T_) + t * T_;
  int lane = lane_id();
  float vals[6];
  float mx = -3.4e38f;
#pragma unroll
  for (int i = 0; i < 6; ++i) {
    int s = lane + i * 32;
    float v = (s < T_ && s <= t) ? row[s] : -3.4e38f;
    vals[i] = v;
    mx = fmaxf(mx, v);
  }
#pragma unroll
  for (int o = 16; o > 0; o >>= 1) mx = fmaxf(mx, __shfl_xor(mx, o, 32));
  float sum = 0.f;
#pragma unroll
  for (int i = 0; i < 6; ++i) {
    int s = lane + i * 32;
    float e = (s < T_ && s <= t) ? expf(vals[i] - mx) : 0.f;
    vals[i] = e;
    sum += e;
  }
#pragma unroll
  for (int o = 16; o > 0; o >>= 1) sum += __shfl_xor(sum, o, 32);
  float inv = 1.f / sum;
#pragma unroll
  for (int i = 0; i < 6; ++i) {
    int s = lane + i * 32;
    if (s < T_) row[s] = vals[i] * inv;
  }
}

// ---------------------------------------------------------------------------
// y = att @ V per head; block per (b,h); f16 V panel in LDS (padded rows).
// ---------------------------------------------------------------------------
__global__ void attn_av(const float* __restrict__ att,
                        const halft* __restrict__ v,
                        halft* __restrict__ y) {
  constexpr int VSTR = 200;  // 192 + 8 pad (bank rotate)
  __shared__ halft vpanel[16 * VSTR];  // [n][s]
  int bh = blockIdx.x;
  int b = bh / H_, h = bh % H_;
  for (int e = threadIdx.x; e < 16 * 192; e += blockDim.x) {
    int n = e / 192, s = e % 192;
    vpanel[n * VSTR + s] =
        (s < T_) ? v[(b * T_ + s) * D_ + h * DH_ + n] : (halft)0.f;
  }
  __syncthreads();

  int lane = lane_id(), wv = wave_id();
  int khalf = (lane >= 16) ? 8 : 0;
  const halft* vp = vpanel + (lane & 15) * VSTR;
#pragma unroll
  for (int i = 0; i < 3; ++i) {
    int mt = wv * 3 + i;  // 12 tiles
    int t = mt * 16 + (lane & 15);
    const float* arow = att + (bh * T_ + (t < T_ ? t : 0)) * T_;
    v8f acc = {};
    for (int k0 = 0; k0 < 192; k0 += 32) {
      v16h a;
#pragma unroll
      for (int j = 0; j < 16; ++j) {
        int s = k0 + ((j < 8) ? (j + khalf) : (j + 8 + khalf));
        a[j] = (t < T_ && s < T_) ? (halft)arow[s] : (halft)0.f;
      }
      v16h bm = frag_from_h16(vp, k0, khalf);
      acc = wmma_f16(a, bm, acc);
    }
    int tt0 = mt * 16 + ((lane >= 16) ? 8 : 0);
    int idx = (b * T_ + tt0) * D_ + h * DH_ + (lane & 15);
    if (mt < 11) {
#pragma unroll
      for (int r = 0; r < 8; ++r) { y[idx] = (halft)acc[r]; idx += D_; }
    } else {
#pragma unroll
      for (int r = 0; r < 8; ++r) {
        if (tt0 + r < T_) y[idx] = (halft)acc[r];
        idx += D_;
      }
    }
  }
}

// ---------------------------------------------------------------------------
// Host-side orchestration.
// ---------------------------------------------------------------------------
extern "C" void kernel_launch(void* const* d_in, const int* in_sizes, int n_in,
                              void* d_out, int out_size, void* d_ws,
                              size_t ws_size, hipStream_t stream) {
  (void)in_sizes; (void)n_in; (void)out_size; (void)ws_size;
  const float* states    = (const float*)d_in[0];
  const int*   actions   = (const int*)  d_in[1];
  const float* rtgs      = (const float*)d_in[2];
  const int*   timesteps = (const int*)  d_in[3];
  const float* conv_w1 = (const float*)d_in[4];
  const float* conv_b1 = (const float*)d_in[5];
  const float* conv_w2 = (const float*)d_in[6];
  const float* conv_b2 = (const float*)d_in[7];
  const float* conv_w3 = (const float*)d_in[8];
  const float* conv_b3 = (const float*)d_in[9];
  const float* enc_w   = (const float*)d_in[10];
  const float* enc_b   = (const float*)d_in[11];
  const float* ret_w   = (const float*)d_in[12];
  const float* ret_b   = (const float*)d_in[13];
  const float* act_emb = (const float*)d_in[14];
  const float* pos_emb = (const float*)d_in[15];
  const float* gpos    = (const float*)d_in[16];
  const float* ln1_g = (const float*)d_in[17];
  const float* ln1_b = (const float*)d_in[18];
  const float* wq = (const float*)d_in[19];
  const float* bq = (const float*)d_in[20];
  const float* wk = (const float*)d_in[21];
  const float* bk = (const float*)d_in[22];
  const float* wv = (const float*)d_in[23];
  const float* bv = (const float*)d_in[24];
  const float* wo = (const float*)d_in[25];
  const float* bo = (const float*)d_in[26];
  const float* ln2_g = (const float*)d_in[27];
  const float* ln2_b = (const float*)d_in[28];
  const float* mlp_w1 = (const float*)d_in[29];
  const float* mlp_b1 = (const float*)d_in[30];
  const float* mlp_w2 = (const float*)d_in[31];
  const float* mlp_b2 = (const float*)d_in[32];
  const float* lnf_g  = (const float*)d_in[33];
  const float* lnf_b  = (const float*)d_in[34];
  const float* head_w = (const float*)d_in[35];

  // Workspace (byte offsets), large regions aliased across phases:
  //   c1h (conv1 out f16) -> att f32 ; c2h -> mlph + y ; c3h -> q/k/v
  uint8_t* w8 = (uint8_t*)d_ws;
  halft* c1h  = (halft*)(w8 + 0);          // 52,428,800 B
  halft* c2h  = (halft*)(w8 + 52428800);   // 21,233,664 B
  halft* c3h  = (halft*)(w8 + 73662464);   // 12,845,056 B
  float* semb = (float*)(w8 + 86507520);   //  1,048,576 B
  float* x    = (float*)(w8 + 87556096);   //  3,129,344 B
  halft* lnb  = (halft*)(w8 + 90685440);   //  1,564,672 B
  float* attw = (float*)(w8 + 0);
  halft* mlph = (halft*)(w8 + 52428800);
  halft* yh   = (halft*)(w8 + 60817408);
  halft* qh   = (halft*)(w8 + 73662464);
  halft* kh   = (halft*)(w8 + 75227136);
  halft* vh   = (halft*)(w8 + 76791808);
  // packed f16 weight panels (LDS images)
  halft* pconv1 = (halft*)(w8 + 92250112);  //  16,896 B
  halft* pconv2 = (halft*)(w8 + 92267008);  //  66,560 B
  halft* pconv3 = (halft*)(w8 + 92333568);  //  74,752 B
  halft* penc   = (halft*)(w8 + 92408320);  // 806,912 B
  halft* pq     = (halft*)(w8 + 93215232);  // 208,896 B (6 layers)
  halft* pk     = (halft*)(w8 + 93424128);  // 208,896 B
  halft* pv     = (halft*)(w8 + 93633024);  // 208,896 B
  halft* po     = (halft*)(w8 + 93841920);  // 208,896 B
  halft* pm1    = (halft*)(w8 + 94050816);  // 835,584 B (6 layers)
  halft* pm2    = (halft*)(w8 + 94886400);  // 798,720 B (6 layers)
  halft* phead  = (halft*)(w8 + 95685120);  //   8,704 B

  const int TPB = 128;  // 4 waves/block
  auto nbw = [](long waves) { return (int)((waves + 3) / 4); };
  auto gblocks = [](int M, int Ntiles) {
    return (((M >> 4) + 15) / 16) * Ntiles;
  };
  auto packT = [&](const float* W, halft* P, int KDIM, int KC, int ldb,
                   int Ncols, int Ntiles) {
    int chunks = KDIM / KC, BSTR = KC + 8;
    int total = Ntiles * chunks * 16 * BSTR;
    pack_weights_t<<<(total + 255) / 256, 256, 0, stream>>>(
        W, P, KC, BSTR, chunks, ldb, Ncols, Ntiles);
  };
  auto packD = [&](const float* W, halft* P, int KDIM, int OC) {
    int BSTR = KDIM + 8;
    int total = (OC / 16) * 16 * BSTR;
    pack_weights_d<<<(total + 255) / 256, 256, 0, stream>>>(
        W, P, KDIM, BSTR, OC);
  };

  // ---- pack all weights into LDS-image panels (one-time, tiny) ----
  packD(conv_w1, pconv1, 256, 32);
  packD(conv_w2, pconv2, 512, 64);
  packD(conv_w3, pconv3, 576, 64);
  packT(enc_w, penc, 3136, 1568, 128, 128, 8);
  const int PQKV = 8 * 16 * 136, PM1 = 32 * 16 * 136, PM2 = 8 * 16 * 520;
  for (int l = 0; l < L_; ++l) {
    packT(wq + (size_t)l * D_ * D_, pq + l * PQKV, 128, 128, 128, 128, 8);
    packT(wk + (size_t)l * D_ * D_, pk + l * PQKV, 128, 128, 128, 128, 8);
    packT(wv + (size_t)l * D_ * D_, pv + l * PQKV, 128, 128, 128, 128, 8);
    packT(wo + (size_t)l * D_ * D_, po + l * PQKV, 128, 128, 128, 128, 8);
    packT(mlp_w1 + (size_t)l * D_ * 4 * D_, pm1 + l * PM1, 128, 128, 512, 512, 32);
    packT(mlp_w2 + (size_t)l * 4 * D_ * D_, pm2 + l * PM2, 512, 512, 128, 128, 8);
  }
  packT(head_w, phead, 128, 128, 18, 18, 2);

  // ---- Nature CNN (implicit-GEMM WMMA) ----
  conv_implicit_gemm<float, 4, 84, 8, 4, 32, 20><<<3200 * 2, TPB, 0, stream>>>(
      states, pconv1, conv_b1, c1h, 2048);
  conv_implicit_gemm<halft, 32, 20, 4, 2, 64, 9><<<648 * 4, TPB, 0, stream>>>(
      c1h, pconv2, conv_b2, c2h, 2048);
  conv_implicit_gemm<halft, 64, 9, 3, 1, 64, 7><<<392 * 4, TPB, 0, stream>>>(
      c2h, pconv3, conv_b3, c3h, 2048);
  // FC 3136->128 + tanh -> f32 state embeddings
  gemm_wmma<2, 3136, 3136><<<gblocks(2048, 8), TPB, 0, stream>>>(
      c3h, penc, enc_b, nullptr, semb, 128, 2048, 128, 128);

  // ---- token assembly ----
  build_tokens<<<(B_ * T_ * D_ + 255) / 256, 256, 0, stream>>>(
      semb, actions, rtgs, timesteps, ret_w, ret_b, act_emb, pos_emb, gpos, x);

  const int rowsBT = B_ * T_;  // 6112
  for (int l = 0; l < L_; ++l) {
    float* attbuf = (l == L_ - 1) ? ((float*)d_out + B_ * K_ * V_) : attw;

    layernorm_kernel<<<nbw(rowsBT), TPB, 0, stream>>>(
        x, ln1_g + l * D_, ln1_b + l * D_, lnb, rowsBT);
    gemm_wmma<0, 128, 128><<<gblocks(rowsBT, 8), TPB, 0, stream>>>(
        lnb, pq + l * PQKV, bq + l * D_, nullptr, qh, 128, rowsBT, 128, 128);
    gemm_wmma<0, 128, 128><<<gblocks(rowsBT, 8), TPB, 0, stream>>>(
        lnb, pk + l * PQKV, bk + l * D_, nullptr, kh, 128, rowsBT, 128, 128);
    gemm_wmma<0, 128, 128><<<gblocks(rowsBT, 8), TPB, 0, stream>>>(
        lnb, pv + l * PQKV, bv + l * D_, nullptr, vh, 128, rowsBT, 128, 128);
    attn_scores<<<nbw(32L * 8 * 12 * 12), TPB, 0, stream>>>(qh, kh, attbuf);
    attn_softmax<<<nbw(32L * 8 * 191), TPB, 0, stream>>>(attbuf);
    attn_av<<<B_ * H_, TPB, 0, stream>>>(attbuf, vh, yh);
    gemm_wmma<4, 128, 128><<<gblocks(rowsBT, 8), TPB, 0, stream>>>(
        yh, po + l * PQKV, bo + l * D_, x, x, 128, rowsBT, 128, 128);
    layernorm_kernel<<<nbw(rowsBT), TPB, 0, stream>>>(
        x, ln2_g + l * D_, ln2_b + l * D_, lnb, rowsBT);
    gemm_wmma<3, 128, 128><<<gblocks(rowsBT, 32), TPB, 0, stream>>>(
        lnb, pm1 + l * PM1, mlp_b1 + l * 4 * D_, nullptr, mlph, 512,
        rowsBT, 512, 512);
    gemm_wmma<4, 512, 512><<<gblocks(rowsBT, 8), TPB, 0, stream>>>(
        mlph, pm2 + l * PM2, mlp_b2 + l * D_, x, x, 128, rowsBT, 128, 128);
  }

  // ---- final LN + head (state tokens only) ----
  layernorm_kernel<<<nbw(rowsBT), TPB, 0, stream>>>(x, lnf_g, lnf_b, lnb, rowsBT);
  gemm_wmma<5, 128, 128><<<gblocks(2048, 2), TPB, 0, stream>>>(
      lnb, phead, nullptr, nullptr, (float*)d_out, 18, 2048, 32, 18);
}
